// MonetMoVDE_88433376625149
// MI455X (gfx1250) — compile-verified, matrix operand order checked by reading, not run on hardware
//
#include <hip/hip_runtime.h>

typedef __attribute__((ext_vector_type(16))) __bf16 v16bf;
typedef __attribute__((ext_vector_type(8)))  __bf16 v8bf;
typedef __attribute__((ext_vector_type(8)))  float  v8f;
typedef __attribute__((ext_vector_type(4)))  int    v4i;

// address-space qualified int4 pointers for the async-to-LDS builtin
typedef __attribute__((address_space(1))) v4i* as1_v4i_p;  // global int4*
typedef __attribute__((address_space(3))) v4i* as3_v4i_p;  // LDS int4*

// ---- problem constants ----
constexpr int Bn = 4, Tn = 2048, Dn = 2048, Hn = 8, En = 512, Mn = 8;
constexpr int Fn   = En * Mn;        // 4096
constexpr int HALFD = Dn / 2;        // 1024
constexpr int NTOK = Bn * Tn;        // 8192
constexpr int KBIG = 2 * Fn + En;    // 8704  (A1 | Y12 | g1s concatenated along K)

// =====================================================================
// convert fp32 -> bf16, with optional column placement into a wider dst
// =====================================================================
__global__ void cvt_bf16(const float* __restrict__ src, __bf16* __restrict__ dst,
                         long long n, int cols, int dstLd, int dstOff) {
  long long i = (long long)blockIdx.x * blockDim.x + threadIdx.x;
  if (i >= n) return;
  int rr = (int)(i / cols);
  int cc = (int)(i % cols);
  dst[(long long)rr * dstLd + dstOff + cc] = (__bf16)src[i];
}

// transpose-convert: src [rowsE, colsC] fp32 -> dst[c*dstLd + dstOff + e]
__global__ void tcvt_bf16(const float* __restrict__ src, __bf16* __restrict__ dst,
                          int rowsE, int colsC, int dstLd, int dstOff) {
  long long n = (long long)rowsE * colsC;
  long long i = (long long)blockIdx.x * blockDim.x + threadIdx.x;
  if (i >= n) return;
  int e = (int)(i / colsC);
  int c = (int)(i % colsC);
  dst[(long long)c * dstLd + dstOff + e] = (__bf16)src[i];
}

// =====================================================================
// WMMA bf16 GEMM:  C[n, c] = sum_k A[n, k] * Bw[c, k]
//   A  : [rows, K]  row-major bf16 (lda)
//   Bw : [cols, K]  row-major bf16 (ldb)   ("NT" gemm, K contiguous)
// EPI==0: c += bias(col); c = relu(c)^2; store bf16 (ldc, coff)
// EPI==1: store fp32 (ldc, coff)
// block = 256 threads (8 waves); tile 128x128x32; wave tile 64x32 (4x2 wmma)
// Double-buffered LDS: async-prefetch tile i+1 while computing tile i;
// s_wait_asynccnt 4 retires only the older tile's 4 in-order async ops.
// =====================================================================
#ifndef USE_ASYNC_LDS
#define USE_ASYNC_LDS 1
#endif
#if USE_ASYNC_LDS && defined(__gfx1250__) && __has_builtin(__builtin_amdgcn_global_load_async_to_lds_b128)
#define HAVE_ASYNC 1
#else
#define HAVE_ASYNC 0
#endif

#if HAVE_ASYNC
// integer round-trip casts: generic->as(1) is value-preserving; generic LDS
// pointer is {aperture_hi32, lds_offset_lo32}, so 32-bit truncation yields the
// exact as(3) offset.
#define ASYNC_CP16(gsrc, ldst)                                                 \
  __builtin_amdgcn_global_load_async_to_lds_b128(                              \
      (as1_v4i_p)(unsigned long long)(gsrc),                                   \
      (as3_v4i_p)(unsigned int)(unsigned long long)(ldst), 0, 0)
#if __has_builtin(__builtin_amdgcn_s_wait_asynccnt)
#define WAIT_ASYNC(n) __builtin_amdgcn_s_wait_asynccnt(n)
#else
#define WAIT_ASYNC(n) asm volatile("s_wait_asynccnt %0" ::"i"(n) : "memory")
#endif
#endif

constexpr int BM = 128, BN = 128, BK = 32;
constexpr int LDT = 40;   // padded LDS row pitch in bf16 (40*2=80B, keeps 16B alignment)

union Frag { v16bf v; v8bf h[2]; };

template <int EPI>
__launch_bounds__(256, 2)
__global__ void wmma_gemm(const __bf16* __restrict__ A, int lda,
                          const __bf16* __restrict__ Bw, int ldb,
                          int K,
                          const float* __restrict__ bias0,
                          const float* __restrict__ bias1, int biasSplit,
                          __bf16* __restrict__ Cbf, float* __restrict__ Cf,
                          int ldc, int coff) {
  __shared__ __bf16 As[2][BM * LDT];
  __shared__ __bf16 Bs[2][BN * LDT];

  const int tid  = threadIdx.x;
  const int lane = tid & 31;
  const int wave = tid >> 5;
  const int wm   = wave & 1;   // 2 row groups of 64
  const int wn   = wave >> 1;  // 4 col groups of 32
  const int r    = lane & 15;
  const int hi   = lane >> 4;
  const int k0   = hi * 8;     // per-lane K base per CDNA5 16-bit A/B layout

  const int rowBase = blockIdx.y * BM;
  const int colBase = blockIdx.x * BN;

  v8f acc[4][2];
#pragma unroll
  for (int a = 0; a < 4; ++a)
#pragma unroll
    for (int b = 0; b < 2; ++b) acc[a][b] = {};

  // tile-load mapping: 512 x 16B chunks per operand tile, 2 chunks/thread
  const int c0 = tid, c1 = tid + 256;
  const int ar0 = c0 >> 2, ak0 = (c0 & 3) * 8;
  const int ar1 = c1 >> 2, ak1 = (c1 & 3) * 8;

  const __bf16* gA0 = A  + (long long)(rowBase + ar0) * lda + ak0;
  const __bf16* gA1 = A  + (long long)(rowBase + ar1) * lda + ak1;
  const __bf16* gB0 = Bw + (long long)(colBase + ar0) * ldb + ak0;
  const __bf16* gB1 = Bw + (long long)(colBase + ar1) * ldb + ak1;

#if HAVE_ASYNC
  // prologue: prefetch tile 0 into buffer 0
  ASYNC_CP16(gA0, &As[0][ar0 * LDT + ak0]);
  ASYNC_CP16(gA1, &As[0][ar1 * LDT + ak1]);
  ASYNC_CP16(gB0, &Bs[0][ar0 * LDT + ak0]);
  ASYNC_CP16(gB1, &Bs[0][ar1 * LDT + ak1]);
#endif

  int it = 0;
  for (int kt = 0; kt < K; kt += BK, ++it) {
    const int cur = it & 1;
#if HAVE_ASYNC
    const int nk = kt + BK;
    if (nk < K) {
      const int nxt = cur ^ 1;
      // safe to overwrite nxt: its last readers passed the trailing barrier
      ASYNC_CP16(gA0 + nk, &As[nxt][ar0 * LDT + ak0]);
      ASYNC_CP16(gA1 + nk, &As[nxt][ar1 * LDT + ak1]);
      ASYNC_CP16(gB0 + nk, &Bs[nxt][ar0 * LDT + ak0]);
      ASYNC_CP16(gB1 + nk, &Bs[nxt][ar1 * LDT + ak1]);
      WAIT_ASYNC(4);  // in-order: tile `cur` landed, tile `nxt` still in flight
    } else {
      WAIT_ASYNC(0);  // last tile: drain
    }
#else
    // synchronous fallback: stage tile kt into the current buffer
    *(v8bf*)&As[cur][ar0 * LDT + ak0] = *(const v8bf*)(gA0 + kt);
    *(v8bf*)&As[cur][ar1 * LDT + ak1] = *(const v8bf*)(gA1 + kt);
    *(v8bf*)&Bs[cur][ar0 * LDT + ak0] = *(const v8bf*)(gB0 + kt);
    *(v8bf*)&Bs[cur][ar1 * LDT + ak1] = *(const v8bf*)(gB1 + kt);
#endif
    __syncthreads();  // tile `cur` visible to all waves

    Frag fa[4], fb[2];
#pragma unroll
    for (int a = 0; a < 4; ++a) {
      const __bf16* p = &As[cur][(wm * 64 + a * 16 + r) * LDT + k0];
      fa[a].h[0] = *(const v8bf*)p;
      fa[a].h[1] = *(const v8bf*)(p + 16);
    }
#pragma unroll
    for (int b = 0; b < 2; ++b) {
      const __bf16* p = &Bs[cur][(wn * 32 + b * 16 + r) * LDT + k0];
      fb[b].h[0] = *(const v8bf*)p;
      fb[b].h[1] = *(const v8bf*)(p + 16);
    }
#pragma unroll
    for (int a = 0; a < 4; ++a)
#pragma unroll
      for (int b = 0; b < 2; ++b)
        acc[a][b] = __builtin_amdgcn_wmma_f32_16x16x32_bf16(
            false, fa[a].v, false, fb[b].v, (short)0, acc[a][b], false, false);
    __syncthreads();  // all waves done reading `cur` before it is refilled
  }

  // epilogue: C layout: lane 0-15 -> N=lane, vgpr v -> M=v (+8 for lanes 16-31)
#pragma unroll
  for (int a = 0; a < 4; ++a) {
#pragma unroll
    for (int b = 0; b < 2; ++b) {
      const int gcol = colBase + wn * 32 + b * 16 + r;
#pragma unroll
      for (int v = 0; v < 8; ++v) {
        const int grow = rowBase + wm * 64 + a * 16 + hi * 8 + v;
        float cv = acc[a][b][v];
        if (EPI == 0) {
          const float bs = (gcol < biasSplit) ? bias0[gcol] : bias1[gcol - biasSplit];
          cv += bs;
          cv = cv > 0.f ? cv * cv : 0.f;  // relu^2
          Cbf[(long long)grow * ldc + coff + gcol] = (__bf16)cv;
        } else {
          Cf[(long long)grow * ldc + coff + gcol] = cv;
        }
      }
    }
  }
}

// =====================================================================
// Per-token mixer: reads g1,g2 and X1/X2 (bf16), builds
//   Z1 = [x1*g1s | Y12 | g1s]   Z2 = [Y21 | x2*g2s | g2s]   (bf16, ld=KBIG)
// where t1[h,m]=sum_i x1[i,m] g1[h,i], Y21[j,m]=sum_h t1[h,m] g2[h,j], etc.
// =====================================================================
__launch_bounds__(256)
__global__ void mixer(const float* __restrict__ g1, const float* __restrict__ g2,
                      const __bf16* __restrict__ Xbf,
                      __bf16* __restrict__ Z1, __bf16* __restrict__ Z2) {
  __shared__ float sg1[Hn * En], sg2[Hn * En];
  __shared__ float s1s[En], s2s[En];
  __shared__ float pt1[4][64], pt2[4][64];
  __shared__ float st1[64], st2[64];

  const int n   = blockIdx.x;
  const int tid = threadIdx.x;
  const float* G1 = g1 + (long long)n * Hn * En;
  const float* G2 = g2 + (long long)n * Hn * En;
  for (int i = tid; i < Hn * En; i += 256) { sg1[i] = G1[i]; sg2[i] = G2[i]; }
  __syncthreads();

  __bf16* z1 = Z1 + (long long)n * KBIG;
  __bf16* z2 = Z2 + (long long)n * KBIG;

  // head-summed gates, also written as the trailing K-block (pairs with b1/b2)
  for (int e = tid; e < En; e += 256) {
    float a = 0.f, b = 0.f;
#pragma unroll
    for (int h = 0; h < Hn; ++h) { a += sg1[h * En + e]; b += sg2[h * En + e]; }
    s1s[e] = a; s2s[e] = b;
    z1[2 * Fn + e] = (__bf16)a;
    z2[2 * Fn + e] = (__bf16)b;
  }

  const __bf16* X1 = Xbf + (long long)n * (2 * Fn);
  const __bf16* X2 = X1 + Fn;

  // t1[h,m], t2[h,m]: 64 thread-groups x 4-way j-split reduction
  {
    const int hm = tid & 63, grp = tid >> 6;
    const int h = hm >> 3, m = hm & 7;
    float a1 = 0.f, a2 = 0.f;
    const int j0 = grp * (En / 4);
    for (int j = j0; j < j0 + En / 4; ++j) {
      a1 += (float)X1[j * Mn + m] * sg1[h * En + j];
      a2 += (float)X2[j * Mn + m] * sg2[h * En + j];
    }
    pt1[grp][hm] = a1; pt2[grp][hm] = a2;
  }
  __syncthreads();
  if (tid < 64) {
    st1[tid] = pt1[0][tid] + pt1[1][tid] + pt1[2][tid] + pt1[3][tid];
    st2[tid] = pt2[0][tid] + pt2[1][tid] + pt2[2][tid] + pt2[3][tid];
  }
  __syncthreads();

  for (int k = tid; k < Fn; k += 256) {
    const int i = k >> 3, m = k & 7;
    const float x1v = (float)X1[k], x2v = (float)X2[k];
    float y12 = 0.f, y21 = 0.f;
#pragma unroll
    for (int h = 0; h < Hn; ++h) {
      y12 += st2[h * 8 + m] * sg1[h * En + i];
      y21 += st1[h * 8 + m] * sg2[h * En + i];
    }
    z1[k]      = (__bf16)(x1v * s1s[i]);
    z1[Fn + k] = (__bf16)y12;
    z2[k]      = (__bf16)y21;
    z2[Fn + k] = (__bf16)(x2v * s2s[i]);
  }
}

// =====================================================================
// host launcher
// =====================================================================
extern "C" void kernel_launch(void* const* d_in, const int* in_sizes, int n_in,
                              void* d_out, int out_size, void* d_ws, size_t ws_size,
                              hipStream_t stream) {
  const float* x    = (const float*)d_in[0];
  const float* g1   = (const float*)d_in[1];
  const float* g2   = (const float*)d_in[2];
  const float* u1_w = (const float*)d_in[3];
  const float* u1_b = (const float*)d_in[4];
  const float* u2_w = (const float*)d_in[5];
  const float* u2_b = (const float*)d_in[6];
  const float* v11  = (const float*)d_in[7];
  const float* v12  = (const float*)d_in[8];
  const float* v21  = (const float*)d_in[9];
  const float* v22  = (const float*)d_in[10];
  const float* b1   = (const float*)d_in[11];
  const float* b2   = (const float*)d_in[12];
  float* out = (float*)d_out;

  char* p = (char*)d_ws;
  auto carve = [&](long long bytes) {
    char* q = p;
    p += (bytes + 255) & ~255LL;
    return (void*)q;
  };
  __bf16* xbf = (__bf16*)carve((long long)NTOK * Dn * 2);       //  33.6 MB
  __bf16* Ubf = (__bf16*)carve((long long)2 * Fn * Dn * 2);     //  33.6 MB
  __bf16* Xbf = (__bf16*)carve((long long)NTOK * 2 * Fn * 2);   // 134.2 MB
  __bf16* Z1  = (__bf16*)carve((long long)NTOK * KBIG * 2);     // 142.6 MB
  __bf16* Z2  = (__bf16*)carve((long long)NTOK * KBIG * 2);     // 142.6 MB
  __bf16* W1  = (__bf16*)carve((long long)HALFD * KBIG * 2);    //  17.8 MB
  __bf16* W2  = (__bf16*)carve((long long)HALFD * KBIG * 2);    //  17.8 MB

  auto cvt = [&](const float* s, __bf16* d, long long n, int cols, int ld, int off) {
    const int thr = 256;
    const long long blocks = (n + thr - 1) / thr;
    cvt_bf16<<<(unsigned)blocks, thr, 0, stream>>>(s, d, n, cols, ld, off);
  };
  // activations and up-projection weights (concat u1;u2 along rows)
  cvt(x,    xbf, (long long)NTOK * Dn, Dn, Dn, 0);
  cvt(u1_w, Ubf,                       (long long)Fn * Dn, Dn, Dn, 0);
  cvt(u2_w, Ubf + (long long)Fn * Dn,  (long long)Fn * Dn, Dn, Dn, 0);
  // down-projection weights, K-concatenated: W1 = [v11 | v12 | b1^T]
  cvt(v11, W1, (long long)HALFD * Fn, Fn, KBIG, 0);
  cvt(v12, W1, (long long)HALFD * Fn, Fn, KBIG, Fn);
  cvt(v21, W2, (long long)HALFD * Fn, Fn, KBIG, 0);
  cvt(v22, W2, (long long)HALFD * Fn, Fn, KBIG, Fn);
  {
    const long long n = (long long)En * HALFD;
    tcvt_bf16<<<(unsigned)((n + 255) / 256), 256, 0, stream>>>(b1, W1, En, HALFD, KBIG, 2 * Fn);
    tcvt_bf16<<<(unsigned)((n + 255) / 256), 256, 0, stream>>>(b2, W2, En, HALFD, KBIG, 2 * Fn);
  }

  // GEMM1: X = relu(x @ [u1;u2]^T + b)^2, bf16 out  [8192 x 8192]
  {
    dim3 grid(2 * Fn / BN, NTOK / BM);
    wmma_gemm<0><<<grid, 256, 0, stream>>>(xbf, Dn, Ubf, Dn, Dn,
                                           u1_b, u2_b, Fn,
                                           Xbf, nullptr, 2 * Fn, 0);
  }

  // per-token gate mixing -> Z1, Z2
  mixer<<<NTOK, 256, 0, stream>>>(g1, g2, Xbf, Z1, Z2);

  // GEMM2: out[:, :1024] = Z1 @ W1^T ; out[:, 1024:] = Z2 @ W2^T  (fp32 out)
  {
    dim3 grid(HALFD / BN, NTOK / BM);
    wmma_gemm<1><<<grid, 256, 0, stream>>>(Z1, KBIG, W1, KBIG, KBIG,
                                           nullptr, nullptr, 0,
                                           nullptr, out, Dn, 0);
    wmma_gemm<1><<<grid, 256, 0, stream>>>(Z2, KBIG, W2, KBIG, KBIG,
                                           nullptr, nullptr, 0,
                                           nullptr, out, Dn, HALFD);
  }
}
